// SAModule_52974126629568
// MI455X (gfx1250) — compile-verified, matrix operand order checked by reading, not run on hardware
//
#include <hip/hip_runtime.h>

#define NPTS 16384
#define S_TOT 4096
#define CIN 64
#define DIN 67
#define KPAD 96
#define H1C 128
#define H2C 128
#define H3C 256
#define KNBR 32
#define CPB 4
#define ROWS 128
#define CAP 256
#define PPT 16   // points per thread in FPS (16 * 1024 = 16384)

typedef __attribute__((ext_vector_type(16))) _Float16 v16h;
typedef __attribute__((ext_vector_type(8)))  float    v8f;

#define WMMA_F16(A, B, C) \
  __builtin_amdgcn_wmma_f32_16x16x32_f16(false, (A), false, (B), (short)0, (C), false, false)

// Load a 16x32 f16 fragment (A-layout; also valid for B when weights are
// stored transposed [col][k]): lane l<16 -> row=rc0+l, K={0..7,16..23}+k0,
// lane l>=16 -> row=rc0+(l&15), K={8..15,24..31}+k0.  ISA 05_wmma 7.12.2.
__device__ __forceinline__ v16h ldFrag(const _Float16* __restrict__ buf,
                                       int rc0, int k0, int ld, int lane) {
  v16h a;
  const _Float16* p = buf + (rc0 + (lane & 15)) * ld + k0 + ((lane >> 4) << 3);
#pragma unroll
  for (int v = 0; v < 8; ++v) {
    int kk = 2 * v + ((v >= 4) ? 8 : 0);
    a[2 * v]     = p[kk];
    a[2 * v + 1] = p[kk + 1];
  }
  return a;
}

// ---------------------------------------------------------------------------
// Kernel 1: farthest point sampling. Serial chain of 4096 argmax steps.
// Distances + coordinates live in VGPRs (16 pts/thread); the min-update is
// fused into the next scan, and the block argmax is one wave-shuffle reduce +
// one ds_max_u64 atomic on a packed (dist_bits<<32 | ~idx) key. Four rotating
// LDS slots let the slot reset overlap => exactly ONE barrier per iteration.
// Positions also staged in LDS (192KB) purely for the winner-coordinate
// broadcast read. Tie-break == jnp.argmax (lowest index).
// ---------------------------------------------------------------------------
__global__ __launch_bounds__(1024) void fps_kernel(
    const float* __restrict__ pos, const int* __restrict__ batch_in,
    int* __restrict__ fpsidx, float* __restrict__ ctrs,
    int* __restrict__ batch_out) {
  __shared__ float px[NPTS], py[NPTS], pz[NPTS];
  __shared__ unsigned long long redbuf[4];
  const int tid = threadIdx.x;
  const int lane = tid & 31;

  float rx[PPT], ry[PPT], rz[PPT], rd[PPT];

  const float x0 = pos[0], y0 = pos[1], z0 = pos[2];
#pragma unroll
  for (int k = 0; k < PPT; ++k) {
    const int i = k * 1024 + tid;
    rx[k] = pos[3 * i];
    ry[k] = pos[3 * i + 1];
    rz[k] = pos[3 * i + 2];
    px[i] = rx[k]; py[i] = ry[k]; pz[i] = rz[k];
    float dx = rx[k] - x0, dy = ry[k] - y0, dz = rz[k] - z0;
    rd[k] = dx * dx + dy * dy + dz * dz;
  }
  if (tid == 0) {
    fpsidx[0] = 0;
    redbuf[0] = 0ull; redbuf[1] = 0ull; redbuf[2] = 0ull; redbuf[3] = 0ull;
  }
  __syncthreads();

  for (int s = 1; s < S_TOT; ++s) {
    // local argmax over register-resident distances (packed order-preserving key)
    unsigned long long best = 0ull;
#pragma unroll
    for (int k = 0; k < PPT; ++k) {
      unsigned long long key =
          ((unsigned long long)__float_as_uint(rd[k]) << 32) |
          (unsigned long long)(0xFFFFFFFFu - (unsigned)(k * 1024 + tid));
      best = key > best ? key : best;
    }
#pragma unroll
    for (int off = 16; off; off >>= 1) {
      unsigned long long o = __shfl_down(best, off);
      best = o > best ? o : best;
    }
    if (lane == 0) atomicMax(&redbuf[s & 3], best);
    if (tid == 0) redbuf[(s + 2) & 3] = 0ull;  // reset slot for iteration s+2
    __syncthreads();
    const unsigned long long win = redbuf[s & 3];
    const int nxt = (int)(0xFFFFFFFFu - (unsigned)(win & 0xFFFFFFFFull));
    if (tid == 0) fpsidx[s] = nxt;
    const float nx = px[nxt], ny = py[nxt], nz = pz[nxt];
    // fused min-update (registers only; next scan sees it, no extra barrier)
#pragma unroll
    for (int k = 0; k < PPT; ++k) {
      float dx = rx[k] - nx, dy = ry[k] - ny, dz = rz[k] - nz;
      float nd = dx * dx + dy * dy + dz * dz;
      rd[k] = nd < rd[k] ? nd : rd[k];
    }
  }
  __syncthreads();
  for (int s = tid; s < S_TOT; s += 1024) {
    int j = fpsidx[s];
    ctrs[3 * s]     = px[j];
    ctrs[3 * s + 1] = py[j];
    ctrs[3 * s + 2] = pz[j];
    batch_out[s] = batch_in[j];
  }
}

// ---------------------------------------------------------------------------
// Kernel 2: ball query — for each center, the set of <=32 nearest points with
// d2 <= r^2 (order irrelevant for the downstream max-pool). One block/center:
// scan 16384 points (L2-resident), push candidates to LDS, 32x argmin select.
// ---------------------------------------------------------------------------
__global__ __launch_bounds__(256) void ballq_kernel(
    const float* __restrict__ pos, const int* __restrict__ fpsidx,
    int* __restrict__ colbuf, unsigned* __restrict__ vmask) {
  __shared__ float cd[CAP];
  __shared__ int   ci[CAP];
  __shared__ int   cnt;
  __shared__ float rv[8];
  __shared__ int   ri[8];
  const int s = blockIdx.x;
  const int tid = threadIdx.x;
  const int lane = tid & 31, wv = tid >> 5;
  const float INF = __builtin_inff();

  if (tid == 0) cnt = 0;
  __syncthreads();
  const int ctr = fpsidx[s];
  const float cx = pos[3 * ctr], cy = pos[3 * ctr + 1], cz = pos[3 * ctr + 2];
  const float R2 = 0.04f;  // r = 0.2
  for (int i = tid; i < NPTS; i += 256) {
    if (i + 512 < NPTS) __builtin_prefetch(pos + 3 * (i + 512), 0, 1);
    float dx = pos[3 * i] - cx, dy = pos[3 * i + 1] - cy, dz = pos[3 * i + 2] - cz;
    float d2 = dx * dx + dy * dy + dz * dz;
    if (d2 <= R2) {
      int slot = atomicAdd(&cnt, 1);
      if (slot < CAP) { cd[slot] = d2; ci[slot] = i; }
    }
  }
  __syncthreads();
  const int m = cnt < CAP ? cnt : CAP;
  unsigned vmacc = 0;
  for (int k = 0; k < KNBR; ++k) {
    float bv = (tid < m) ? cd[tid] : INF;
    int   bi = tid;
#pragma unroll
    for (int off = 16; off; off >>= 1) {
      float v2 = __shfl_down(bv, off);
      int   i2 = __shfl_down(bi, off);
      if (v2 < bv || (v2 == bv && i2 < bi)) { bv = v2; bi = i2; }
    }
    if (lane == 0) { rv[wv] = bv; ri[wv] = bi; }
    __syncthreads();
    if (tid == 0) {
      float mv = rv[0]; int mi = ri[0];
      for (int q = 1; q < 8; ++q)
        if (rv[q] < mv || (rv[q] == mv && ri[q] < mi)) { mv = rv[q]; mi = ri[q]; }
      if (k < m) {
        colbuf[s * KNBR + k] = ci[mi];
        cd[mi] = INF;
        vmacc |= (1u << k);
      } else {
        colbuf[s * KNBR + k] = 0;
      }
    }
    __syncthreads();
  }
  if (tid == 0) vmask[s] = vmacc;
}

// ---------------------------------------------------------------------------
// Kernel 2.5: one-shot weight transpose + f32->f16 conversion into workspace
// (was previously redone by each of the 1024 MLP blocks).
// ---------------------------------------------------------------------------
__global__ __launch_bounds__(256) void prep_weights(
    const float* __restrict__ W1, const float* __restrict__ W2,
    const float* __restrict__ W3, _Float16* __restrict__ w1t,
    _Float16* __restrict__ w2t, _Float16* __restrict__ w3t) {
  const int g = blockIdx.x * 256 + threadIdx.x;
  if (g < H1C * KPAD) {
    int n = g / KPAD, k = g % KPAD;
    w1t[g] = (_Float16)((k < DIN) ? W1[k * H1C + n] : 0.0f);
  } else if (g < H1C * KPAD + H2C * H1C) {
    int e = g - H1C * KPAD;
    int n = e / H1C, k = e % H1C;
    w2t[e] = (_Float16)W2[k * H2C + n];
  } else if (g < H1C * KPAD + H2C * H1C + H3C * H2C) {
    int e = g - (H1C * KPAD + H2C * H1C);
    int n = e / H2C, k = e % H2C;
    w3t[e] = (_Float16)W3[k * H3C + n];
  }
}

// ---------------------------------------------------------------------------
// Kernel 3: fused gather + 3-layer MLP (WMMA f16->f32) + masked max-pool.
// Block = 4 centers = 128 rows. Pre-transposed f16 weights bulk-copied from
// workspace (L2-resident) into LDS as uint4; wave w owns N-tile w, B
// fragments held in registers across M-tiles. Layer-3 max fused into the
// WMMA epilogue (no h3 materialization).
// LDS: 120KB weights + 88KB activations = 208KB (<320KB/WGP).
// ---------------------------------------------------------------------------
__global__ __launch_bounds__(256) void mlp_kernel(
    const float* __restrict__ x, const float* __restrict__ pos,
    const _Float16* __restrict__ wts,  // w1t | w2t | w3t, contiguous
    const float* __restrict__ b1, const float* __restrict__ b2,
    const float* __restrict__ b3,
    const int* __restrict__ fpsidx, const int* __restrict__ colbuf,
    const unsigned* __restrict__ vmask, float* __restrict__ out) {
  __shared__ __align__(16) _Float16 wsh[H1C * KPAD + H2C * H1C + H3C * H2C];
  __shared__ __align__(16) _Float16 bIn[ROWS * KPAD];
  __shared__ __align__(16) _Float16 bH1[ROWS * H1C];
  __shared__ __align__(16) _Float16 bH2[ROWS * H2C];
  _Float16* const w1t = wsh;
  _Float16* const w2t = wsh + H1C * KPAD;
  _Float16* const w3t = wsh + H1C * KPAD + H2C * H1C;

  const int tid = threadIdx.x;
  const int lane = tid & 31, w = tid >> 5;
  const int c0 = blockIdx.x * CPB;
  const float NEGINF = -__builtin_inff();

  // --- bulk-copy staged f16 weights into LDS (16B per transfer) ---
  {
    const uint4* src = (const uint4*)wts;
    uint4* dst = (uint4*)wsh;
    const int nvec = (H1C * KPAD + H2C * H1C + H3C * H2C) / 8;  // 7680
    __builtin_prefetch(src + tid, 0, 1);
    for (int e = tid; e < nvec; e += 256) dst[e] = src[e];
  }

  // --- gather input rows: [x_j , pos_j - pos_i , 0-pad] ---
  {
    const int r = tid >> 1, half = tid & 1;
    const int c = r >> 5, j = r & 31;
    const int sg = c0 + c;
    const int ni = colbuf[sg * KNBR + j];
    const bool vld = (vmask[sg] >> j) & 1u;
    const int ctr = fpsidx[sg];
    const float cxp = pos[3 * ctr], cyp = pos[3 * ctr + 1], czp = pos[3 * ctr + 2];
    for (int kk = 0; kk < 48; ++kk) {
      int k = half * 48 + kk;
      float v = 0.0f;
      if (vld) {
        if (k < CIN)            v = x[ni * CIN + k];
        else if (k == CIN)      v = pos[3 * ni]     - cxp;
        else if (k == CIN + 1)  v = pos[3 * ni + 1] - cyp;
        else if (k == CIN + 2)  v = pos[3 * ni + 2] - czp;
      }
      bIn[r * KPAD + k] = (_Float16)v;
    }
  }
  __syncthreads();

  // --- layer 1: [128 x 96] @ [96 x 128], relu ---
  {
    v16h B0 = ldFrag(w1t, w * 16, 0,  KPAD, lane);
    v16h B1 = ldFrag(w1t, w * 16, 32, KPAD, lane);
    v16h B2 = ldFrag(w1t, w * 16, 64, KPAD, lane);
    const float bb = b1[w * 16 + (lane & 15)];
    for (int mt = 0; mt < 8; ++mt) {
      v8f acc = {};
      v16h A = ldFrag(bIn, mt * 16, 0, KPAD, lane);
      acc = WMMA_F16(A, B0, acc);
      A = ldFrag(bIn, mt * 16, 32, KPAD, lane);
      acc = WMMA_F16(A, B1, acc);
      A = ldFrag(bIn, mt * 16, 64, KPAD, lane);
      acc = WMMA_F16(A, B2, acc);
      const int rbase = mt * 16 + ((lane >> 4) << 3);
      const int ccol = w * 16 + (lane & 15);
#pragma unroll
      for (int i = 0; i < 8; ++i) {
        float hv = acc[i] + bb;
        hv = hv > 0.0f ? hv : 0.0f;
        bH1[(rbase + i) * H1C + ccol] = (_Float16)hv;
      }
    }
  }
  __syncthreads();

  // --- layer 2: [128 x 128] @ [128 x 128], relu ---
  {
    v16h B0 = ldFrag(w2t, w * 16, 0,  H1C, lane);
    v16h B1 = ldFrag(w2t, w * 16, 32, H1C, lane);
    v16h B2 = ldFrag(w2t, w * 16, 64, H1C, lane);
    v16h B3 = ldFrag(w2t, w * 16, 96, H1C, lane);
    const float bb = b2[w * 16 + (lane & 15)];
    for (int mt = 0; mt < 8; ++mt) {
      v8f acc = {};
      v16h A = ldFrag(bH1, mt * 16, 0, H1C, lane);
      acc = WMMA_F16(A, B0, acc);
      A = ldFrag(bH1, mt * 16, 32, H1C, lane);
      acc = WMMA_F16(A, B1, acc);
      A = ldFrag(bH1, mt * 16, 64, H1C, lane);
      acc = WMMA_F16(A, B2, acc);
      A = ldFrag(bH1, mt * 16, 96, H1C, lane);
      acc = WMMA_F16(A, B3, acc);
      const int rbase = mt * 16 + ((lane >> 4) << 3);
      const int ccol = w * 16 + (lane & 15);
#pragma unroll
      for (int i = 0; i < 8; ++i) {
        float hv = acc[i] + bb;
        hv = hv > 0.0f ? hv : 0.0f;
        bH2[(rbase + i) * H2C + ccol] = (_Float16)hv;
      }
    }
  }
  __syncthreads();

  // --- layer 3: [128 x 128] @ [128 x 256], bias, validity mask, fused max ---
  for (int half = 0; half < 2; ++half) {
    const int n0 = (w + 8 * half) * 16;
    v16h B0 = ldFrag(w3t, n0, 0,  H2C, lane);
    v16h B1 = ldFrag(w3t, n0, 32, H2C, lane);
    v16h B2 = ldFrag(w3t, n0, 64, H2C, lane);
    v16h B3 = ldFrag(w3t, n0, 96, H2C, lane);
    const float bb = b3[n0 + (lane & 15)];
    for (int c = 0; c < CPB; ++c) {
      const unsigned vm = vmask[c0 + c];
      float pm = NEGINF;
#pragma unroll
      for (int h2t = 0; h2t < 2; ++h2t) {
        const int mt = 2 * c + h2t;
        v8f acc = {};
        v16h A = ldFrag(bH2, mt * 16, 0, H2C, lane);
        acc = WMMA_F16(A, B0, acc);
        A = ldFrag(bH2, mt * 16, 32, H2C, lane);
        acc = WMMA_F16(A, B1, acc);
        A = ldFrag(bH2, mt * 16, 64, H2C, lane);
        acc = WMMA_F16(A, B2, acc);
        A = ldFrag(bH2, mt * 16, 96, H2C, lane);
        acc = WMMA_F16(A, B3, acc);
        const int rl0 = h2t * 16 + ((lane >> 4) << 3);  // neighbor slot base
#pragma unroll
        for (int i = 0; i < 8; ++i) {
          const int slot = rl0 + i;
          float hv = acc[i] + bb;
          hv = ((vm >> slot) & 1u) ? hv : NEGINF;
          pm = pm > hv ? pm : hv;
        }
      }
      pm = fmaxf(pm, __shfl_xor(pm, 16));
      if (lane < 16) out[(c0 + c) * H3C + n0 + lane] = pm;
    }
  }
}

// ---------------------------------------------------------------------------
extern "C" void kernel_launch(void* const* d_in, const int* in_sizes, int n_in,
                              void* d_out, int out_size, void* d_ws, size_t ws_size,
                              hipStream_t stream) {
  const float* x     = (const float*)d_in[0];
  const float* pos   = (const float*)d_in[1];
  const int*   batch = (const int*)d_in[2];
  const float* W1    = (const float*)d_in[3];
  const float* b1    = (const float*)d_in[4];
  const float* W2    = (const float*)d_in[5];
  const float* b2    = (const float*)d_in[6];
  const float* W3    = (const float*)d_in[7];
  const float* b3    = (const float*)d_in[8];

  float* out       = (float*)d_out;              // [4096, 256]
  float* ctrs      = out + S_TOT * H3C;          // [4096, 3]
  int*   batch_out = (int*)(ctrs + S_TOT * 3);   // [4096]

  char* ws = (char*)d_ws;
  int*      fpsidx = (int*)ws;                                        // 16 KB
  int*      colbuf = (int*)(ws + 4 * S_TOT);                          // 512 KB
  unsigned* vmaskp = (unsigned*)(ws + 4 * S_TOT + 4 * S_TOT * KNBR);  // 16 KB
  _Float16* wtsbuf = (_Float16*)(ws + 4 * S_TOT + 4 * S_TOT * KNBR + 4 * S_TOT);
  _Float16* w1t = wtsbuf;
  _Float16* w2t = wtsbuf + H1C * KPAD;
  _Float16* w3t = wtsbuf + H1C * KPAD + H2C * H1C;

  const int wtot = H1C * KPAD + H2C * H1C + H3C * H2C;  // 61440
  fps_kernel<<<1, 1024, 0, stream>>>(pos, batch, fpsidx, ctrs, batch_out);
  prep_weights<<<(wtot + 255) / 256, 256, 0, stream>>>(W1, W2, W3, w1t, w2t, w3t);
  ballq_kernel<<<S_TOT, 256, 0, stream>>>(pos, fpsidx, colbuf, vmaskp);
  mlp_kernel<<<S_TOT / CPB, 256, 0, stream>>>(x, pos, wtsbuf, b1, b2, b3,
                                              fpsidx, colbuf, vmaskp, out);
}
